// GCN_33741263077612
// MI455X (gfx1250) — matrix operand order repaired and checked by reading, not compiled, not run on hardware
//
#include <hip/hip_runtime.h>

// ---------------------------------------------------------------------------
// GCN forward for MI455X (gfx1250, wave32, WMMA, async-LDS staging).
//   x1 = relu(adj @ (x  @ W1) + b1)
//   x2 = relu(adj @ (x1 @ W2) + b2)
//   x3 =       adj @ (x2 @ W3) + b3
//   out = log_softmax([x1|x2|x3] @ Wl + bl)
//
// bf16-in / fp32-accumulate on the matrix core; one-shot cast kernels; weight
// GEMMs emit transposed bf16 so the big adj GEMMs stage tiles as contiguous
// 16-byte copies; gfx1250 async global->LDS staging with double buffering.
// Wave tile mapping: 2 m-tiles x (BN/64) n-tiles per wave -> A and B fragments
// are each reused, 12 ds_load_b128 per 8 WMMAs per 32-K step.
// ---------------------------------------------------------------------------

typedef __attribute__((ext_vector_type(16))) __bf16 v16bf;
typedef __attribute__((ext_vector_type(8)))  float  v8f;
typedef int v4i __attribute__((ext_vector_type(4)));

struct alignas(16) U128 { unsigned int u[4]; };
struct U256 { U128 a, b; };   // 32 bytes == one v16bf WMMA operand

static __device__ __forceinline__ __bf16 f2bf(float f) {
    unsigned u = __builtin_bit_cast(unsigned, f);
    unsigned r = u + 0x7FFFu + ((u >> 16) & 1u);   // round-to-nearest-even
    unsigned short h = (unsigned short)(r >> 16);
    return __builtin_bit_cast(__bf16, h);
}
static __device__ __forceinline__ unsigned pack2(float a, float b) {
    unsigned la = (unsigned)__builtin_bit_cast(unsigned short, f2bf(a));
    unsigned lb = (unsigned)__builtin_bit_cast(unsigned short, f2bf(b));
    return la | (lb << 16);
}

// ---- gfx1250 async global->LDS copy (guarded), else plain uint4 copy ------
#if defined(__has_builtin)
#  if __has_builtin(__builtin_amdgcn_global_load_async_to_lds_b128)
#    define GCN_ASYNC_LDS 1
#  endif
#endif

#ifdef GCN_ASYNC_LDS
typedef __attribute__((address_space(1))) v4i GV4;   // global int4
typedef __attribute__((address_space(3))) v4i LV4;   // LDS int4
#endif

static __device__ __forceinline__ void copy16_g2l(const void* g, void* l) {
#ifdef GCN_ASYNC_LDS
    __builtin_amdgcn_global_load_async_to_lds_b128(
        (GV4*)(const_cast<void*>(g)), (LV4*)(l), 0, 0);
#else
    *(uint4*)l = *(const uint4*)g;
#endif
}
static __device__ __forceinline__ void wait_stage() {
#ifdef GCN_ASYNC_LDS
#  if __has_builtin(__builtin_amdgcn_s_wait_asynccnt)
    __builtin_amdgcn_s_wait_asynccnt(0);
#  else
    asm volatile("s_wait_asynccnt 0x0" ::: "memory");
#  endif
#endif
}

// ---------------------------------------------------------------------------
// C[M,N] = A[M,K] @ Bt[N,K]^T   (+bias, +relu)      A row-major, Bt row-major.
// AF32:  A is fp32 (converted while staging); else A is bf16.
// TRANSC: write C as transposed bf16 Ct[N][M]; else fp32 C[M][N].
// Block: 256 threads = 8 wave32, covers BM=64 rows x BN cols.
// Wave w: m-tiles {2*(w>>2), 2*(w>>2)+1}, n-tiles (w&3)*NT + j, NT = BN/64.
// Requires M%64==0, N%BN==0, K%32==0.
// ---------------------------------------------------------------------------
template<int BN, bool AF32, bool TRANSC, bool BIAS, bool RELU>
__global__ __launch_bounds__(256)
void gemm_wmma(const void* __restrict__ Av, const __bf16* __restrict__ Bt,
               const float* __restrict__ bias, void* __restrict__ Cv,
               int M, int N, int K) {
    constexpr int BM = 64;
    constexpr int NT = BN / 64;                 // n-tiles per wave (4 or 2)
    __shared__ __align__(16) __bf16 sA[2][BM][40];
    __shared__ __align__(16) __bf16 sB[2][BN][40];

    const int tid  = threadIdx.x;
    const int lane = tid & 31;
    const int wv   = tid >> 5;
    const int lo   = lane & 15;
    const int hi   = lane >> 4;
    const int mt2  = wv >> 2;                   // 0..1 : m-tile pair
    const int nq   = wv & 3;                    // 0..3 : n-tile quad

    const int row0 = blockIdx.y * BM;
    const int col0 = blockIdx.x * BN;

    const __bf16* Ab = (const __bf16*)Av;
    const float*  Af = (const float*)Av;

    // staging coords: thread copies 8 bf16 (16B) per chunk
    const int sr = tid >> 2;                    // 0..63
    const int sc = (tid & 3) * 8;               // 0,8,16,24

    auto stage = [&](int kk, int p) {
        if constexpr (AF32) {
            const float* pa = Af + (size_t)(row0 + sr) * (size_t)K + (size_t)(kk + sc);
            float4 f0 = *(const float4*)pa;
            float4 f1 = *(const float4*)(pa + 4);
            uint4 q;
            q.x = pack2(f0.x, f0.y); q.y = pack2(f0.z, f0.w);
            q.z = pack2(f1.x, f1.y); q.w = pack2(f1.z, f1.w);
            *(uint4*)&sA[p][sr][sc] = q;
        } else {
            copy16_g2l(Ab + (size_t)(row0 + sr) * (size_t)K + (size_t)(kk + sc),
                       &sA[p][sr][sc]);
        }
#pragma unroll
        for (int i = 0; i < BN / 64; ++i) {
            int c = sr + i * 64;
            copy16_g2l(Bt + (size_t)(col0 + c) * (size_t)K + (size_t)(kk + sc),
                       &sB[p][c][sc]);
        }
    };

    v8f acc[2][NT] = {};

    stage(0, 0);
    int p = 0;
    for (int kk = 0; kk < K; kk += 32) {
        wait_stage();
        __syncthreads();                         // staged data visible; prev buffer free
        if (kk + 32 < K) stage(kk + 32, p ^ 1);  // overlap next-tile staging with WMMA

        // hoist all fragment loads so the scheduler batches ds_load_b128s
        v16bf af[2];
#pragma unroll
        for (int i = 0; i < 2; ++i) {
            U256 ta;
            ta.a = *(const U128*)&sA[p][(mt2 * 2 + i) * 16 + lo][hi * 8];
            ta.b = *(const U128*)&sA[p][(mt2 * 2 + i) * 16 + lo][16 + hi * 8];
            af[i] = __builtin_bit_cast(v16bf, ta);
        }
        v16bf bf[NT];
#pragma unroll
        for (int j = 0; j < NT; ++j) {
            int ct = (nq * NT + j) * 16 + lo;
            U256 tb;
            tb.a = *(const U128*)&sB[p][ct][hi * 8];
            tb.b = *(const U128*)&sB[p][ct][16 + hi * 8];
            bf[j] = __builtin_bit_cast(v16bf, tb);
        }
#pragma unroll
        for (int i = 0; i < 2; ++i)
#pragma unroll
            for (int j = 0; j < NT; ++j)
                acc[i][j] = __builtin_amdgcn_wmma_f32_16x16x32_bf16(
                    false, af[i], false, bf[j], (short)0, acc[i][j], false, false);
        p ^= 1;
    }

    if constexpr (TRANSC) {                      // Ct[N][M], bf16, 16B per tile-lane
        __bf16* Ct = (__bf16*)Cv;
#pragma unroll
        for (int i = 0; i < 2; ++i) {
#pragma unroll
            for (int j = 0; j < NT; ++j) {
                int n  = col0 + (nq * NT + j) * 16 + lo;
                int m0 = row0 + (mt2 * 2 + i) * 16 + hi * 8;
                uint4 q;
                q.x = pack2(acc[i][j][0], acc[i][j][1]);
                q.y = pack2(acc[i][j][2], acc[i][j][3]);
                q.z = pack2(acc[i][j][4], acc[i][j][5]);
                q.w = pack2(acc[i][j][6], acc[i][j][7]);
                *(uint4*)(Ct + (size_t)n * (size_t)M + (size_t)m0) = q;
            }
        }
    } else {
        float* C = (float*)Cv;
#pragma unroll
        for (int i = 0; i < 2; ++i) {
#pragma unroll
            for (int j = 0; j < NT; ++j) {
                int n = col0 + (nq * NT + j) * 16 + lo;
                float bv = 0.0f;
                if constexpr (BIAS) bv = bias[n];
#pragma unroll
                for (int r = 0; r < 8; ++r) {
                    int m = row0 + (mt2 * 2 + i) * 16 + hi * 8 + r;
                    float v = acc[i][j][r] + bv;
                    if constexpr (RELU) v = fmaxf(v, 0.0f);
                    C[(size_t)m * (size_t)N + (size_t)n] = v;
                }
            }
        }
    }
}

// ---- one-shot casts -------------------------------------------------------
__global__ __launch_bounds__(256)
void cast_bf16_kernel(const float* __restrict__ src, __bf16* __restrict__ dst) {
    size_t i = ((size_t)blockIdx.x * 256 + threadIdx.x) * 8;
    float4 a = *(const float4*)(src + i);
    float4 b = *(const float4*)(src + i + 4);
    uint4 q;
    q.x = pack2(a.x, a.y); q.y = pack2(a.z, a.w);
    q.z = pack2(b.x, b.y); q.w = pack2(b.z, b.w);
    *(uint4*)(dst + i) = q;
}

// W[K][N] fp32 -> Wt[N][K] bf16  (tiny matrices; coalesced on output)
__global__ __launch_bounds__(256)
void cast_transpose_bf16(const float* __restrict__ W, __bf16* __restrict__ Wt,
                         int K, int N) {
    int idx = blockIdx.x * 256 + threadIdx.x;   // over N*K outputs
    int n = idx / K, k = idx % K;
    Wt[idx] = f2bf(W[(size_t)k * (size_t)N + (size_t)n]);
}

// logits = [X1|X2|X3] @ Wl + bl ; out = log_softmax(logits)  (0.1% of FLOPs)
__global__ __launch_bounds__(256)
void logits_logsoftmax(const float* __restrict__ X1, const float* __restrict__ X2,
                       const float* __restrict__ X3, const float* __restrict__ Wl,
                       const float* __restrict__ bl, float* __restrict__ out) {
    const int row = blockIdx.x * blockDim.x + threadIdx.x;
    float acc[16];
#pragma unroll
    for (int c = 0; c < 16; ++c) acc[c] = bl[c];

    const float* x1 = X1 + (size_t)row * 256;
    for (int k = 0; k < 256; ++k) {
        float xv = x1[k];
        const float* w = Wl + (size_t)k * 16;
#pragma unroll
        for (int c = 0; c < 16; ++c) acc[c] = fmaf(xv, w[c], acc[c]);
    }
    const float* x2 = X2 + (size_t)row * 256;
    for (int k = 0; k < 256; ++k) {
        float xv = x2[k];
        const float* w = Wl + (size_t)(256 + k) * 16;
#pragma unroll
        for (int c = 0; c < 16; ++c) acc[c] = fmaf(xv, w[c], acc[c]);
    }
    const float* x3 = X3 + (size_t)row * 128;
    for (int k = 0; k < 128; ++k) {
        float xv = x3[k];
        const float* w = Wl + (size_t)(512 + k) * 16;
#pragma unroll
        for (int c = 0; c < 16; ++c) acc[c] = fmaf(xv, w[c], acc[c]);
    }

    float m = acc[0];
#pragma unroll
    for (int c = 1; c < 16; ++c) m = fmaxf(m, acc[c]);
    float s = 0.0f;
#pragma unroll
    for (int c = 0; c < 16; ++c) s += expf(acc[c] - m);
    float ls = logf(s);
#pragma unroll
    for (int c = 0; c < 16; ++c) out[(size_t)row * 16 + c] = acc[c] - m - ls;
}

extern "C" void kernel_launch(void* const* d_in, const int* in_sizes, int n_in,
                              void* d_out, int out_size, void* d_ws, size_t ws_size,
                              hipStream_t stream) {
    (void)in_sizes; (void)n_in; (void)out_size;
    constexpr int N  = 16384;
    constexpr int NF = 512;
    constexpr int NH = 256;
    constexpr int NO = 128;

    const float* x   = (const float*)d_in[0];
    const float* adj = (const float*)d_in[1];
    const float* W1  = (const float*)d_in[2];
    const float* b1  = (const float*)d_in[3];
    const float* W2  = (const float*)d_in[4];
    const float* b2  = (const float*)d_in[5];
    const float* W3  = (const float*)d_in[6];
    const float* b3  = (const float*)d_in[7];
    const float* Wl  = (const float*)d_in[8];
    const float* bl  = (const float*)d_in[9];
    float* out = (float*)d_out;

    char* ws = (char*)d_ws;
    size_t off = 0;
    auto alloc = [&](size_t bytes) { char* p = ws + off; off += (bytes + 255) & ~(size_t)255; return p; };

    float*  X1f = (float*)alloc((size_t)N * NH * 4);       // 16 MB
    float*  X2f = (float*)alloc((size_t)N * NH * 4);       // 16 MB
    float*  X3f = (float*)alloc((size_t)N * NO * 4);       //  8 MB
    __bf16* X1b = (__bf16*)alloc((size_t)N * NH * 2);      //  8 MB
    __bf16* X2b = (__bf16*)alloc((size_t)N * NH * 2);      //  8 MB
    __bf16* xb  = (__bf16*)alloc((size_t)N * NF * 2);      // 16 MB
    __bf16* Tt  = (__bf16*)alloc((size_t)NH * N * 2);      //  8 MB  (Tt[N][K])
    __bf16* W1t = (__bf16*)alloc((size_t)NF * NH * 2);
    __bf16* W2t = (__bf16*)alloc((size_t)NH * NH * 2);
    __bf16* W3t = (__bf16*)alloc((size_t)NH * NO * 2);
    const size_t adjBytes = (size_t)N * N * 2;             // 512 MB
    const bool useAdjB = (off + adjBytes) <= ws_size;
    __bf16* adjb = useAdjB ? (__bf16*)alloc(adjBytes) : nullptr;

    dim3 blk(256);
    dim3 gH(1, N / 64);    // BN=256 path
    dim3 gO(1, N / 64);    // BN=128 path

    // -- one-shot bf16 casts --
    cast_bf16_kernel<<<(N * NF) / 2048, blk, 0, stream>>>(x, xb);
    cast_transpose_bf16<<<(NF * NH) / 256, blk, 0, stream>>>(W1, W1t, NF, NH);
    cast_transpose_bf16<<<(NH * NH) / 256, blk, 0, stream>>>(W2, W2t, NH, NH);
    cast_transpose_bf16<<<(NH * NO) / 256, blk, 0, stream>>>(W3, W3t, NH, NO);
    if (useAdjB)
        cast_bf16_kernel<<<((size_t)N * N) / 2048, blk, 0, stream>>>(adj, adjb);

    // -- layer 1 --
    gemm_wmma<256, false, true, false, false><<<gH, blk, 0, stream>>>(xb, W1t, nullptr, Tt, N, NH, NF);
    if (useAdjB)
        gemm_wmma<256, false, false, true, true><<<gH, blk, 0, stream>>>(adjb, Tt, b1, X1f, N, NH, N);
    else
        gemm_wmma<256, true,  false, true, true><<<gH, blk, 0, stream>>>(adj,  Tt, b1, X1f, N, NH, N);
    cast_bf16_kernel<<<(N * NH) / 2048, blk, 0, stream>>>(X1f, X1b);

    // -- layer 2 --
    gemm_wmma<256, false, true, false, false><<<gH, blk, 0, stream>>>(X1b, W2t, nullptr, Tt, N, NH, NH);
    if (useAdjB)
        gemm_wmma<256, false, false, true, true><<<gH, blk, 0, stream>>>(adjb, Tt, b2, X2f, N, NH, N);
    else
        gemm_wmma<256, true,  false, true, true><<<gH, blk, 0, stream>>>(adj,  Tt, b2, X2f, N, NH, N);
    cast_bf16_kernel<<<(N * NH) / 2048, blk, 0, stream>>>(X2f, X2b);

    // -- layer 3 --
    gemm_wmma<128, false, true, false, false><<<gO, blk, 0, stream>>>(X2b, W3t, nullptr, Tt, N, NO, NH);
    if (useAdjB)
        gemm_wmma<128, false, false, true, false><<<gO, blk, 0, stream>>>(adjb, Tt, b3, X3f, N, NO, N);
    else
        gemm_wmma<128, true,  false, true, false><<<gO, blk, 0, stream>>>(adj,  Tt, b3, X3f, N, NO, N);

    // -- head --
    logits_logsoftmax<<<N / 256, blk, 0, stream>>>(X1f, X2f, X3f, Wl, bl, out);
}
